// RoPECausalAttention_29137058136692
// MI455X (gfx1250) — matrix operand order repaired and checked by reading, not compiled, not run on hardware
//
#include <hip/hip_runtime.h>
#include <hip/hip_bf16.h>

// ---------------------------------------------------------------------------
// RoPE causal attention for MI455X (gfx1250, wave32, WMMA).
// B=2, C=2048, D=1024, H=16, dh=64.
// GEMMs: LDS-tiled 128x128 block tiles, bf16 WMMA, async global->LDS staging
// (ASYNCcnt) via the gfx1250 async-to-LDS instructions.
// ---------------------------------------------------------------------------

#define BB   2
#define CC   2048
#define DM   1024
#define HH   16
#define DH   64
#define ROWS (BB * CC)      // 4096
#define QKVN (3 * DM)       // 3072

typedef __attribute__((ext_vector_type(16))) __bf16 v16bf;
typedef __attribute__((ext_vector_type(8)))  float  v8f;
typedef __attribute__((ext_vector_type(4)))  int    v4i;

union FragAB {
    v16bf v;
    uint4 u[2];
};

#if __has_builtin(__builtin_amdgcn_global_load_async_to_lds_b128)
#define ASYNC_LDS 1
#endif

// Copy 8 bf16 (16 B) global -> LDS. Async DMA path on CDNA5 if available.
__device__ __forceinline__ void stage16(const __bf16* g, __bf16* l) {
#ifdef ASYNC_LDS
    __builtin_amdgcn_global_load_async_to_lds_b128(
        (__attribute__((address_space(1))) v4i*)g,
        (__attribute__((address_space(3))) v4i*)l, 0, 0);
#else
    *reinterpret_cast<uint4*>(l) = *reinterpret_cast<const uint4*>(g);
#endif
}

__device__ __forceinline__ void wait_stage() {
#ifdef ASYNC_LDS
#if __has_builtin(__builtin_amdgcn_s_wait_asynccnt)
    __builtin_amdgcn_s_wait_asynccnt(0);
#else
    asm volatile("s_wait_asynccnt 0x0" ::: "memory");
#endif
#endif
}

// A-fragment (16x32 bf16, MxK): lanes 0-15 hold row M=lane, K=0..7 & 16..23;
// lanes 16-31 hold row M=lane-16, K=8..15 & 24..31. Each chunk is 16B.
__device__ __forceinline__ v16bf load_a_frag(const __bf16* row_k0, int lane) {
    FragAB f;
    const int off = (lane < 16) ? 0 : 8;
    f.u[0] = *reinterpret_cast<const uint4*>(row_k0 + off);
    f.u[1] = *reinterpret_cast<const uint4*>(row_k0 + off + 16);
    return f.v;
}

// B-fragment (32x16 bf16, KxN): lanes 0-15 hold col N=lane, K=0..7 & 16..23;
// lanes 16-31 hold col N=lane-16, K=8..15 & 24..31. K is contiguous in memory here.
__device__ __forceinline__ v16bf load_b_frag(const __bf16* col_k0, int lane) {
    FragAB f;
    const __bf16* p = col_k0 + ((lane < 16) ? 0 : 8);
    f.u[0] = *reinterpret_cast<const uint4*>(p);
    f.u[1] = *reinterpret_cast<const uint4*>(p + 16);
    return f.v;
}

__device__ __forceinline__ v8f wmma_bf16(v16bf a, v16bf b, v8f c) {
    return __builtin_amdgcn_wmma_f32_16x16x32_bf16(
        /*neg_a=*/false, a, /*neg_b=*/false, b,
        /*c_mod=*/(short)0, c, /*reuse_a=*/false, /*reuse_b=*/false);
}

// ---------------------------------------------------------------------------
// f32 -> bf16 cast
// ---------------------------------------------------------------------------
__global__ void cast_f32_to_bf16(const float* __restrict__ in,
                                 __bf16* __restrict__ out, int n) {
    int i = blockIdx.x * blockDim.x + threadIdx.x;
    if (i < n) out[i] = (__bf16)in[i];
}

// ---------------------------------------------------------------------------
// LDS-tiled GEMM: C[M,N] = A[M,K] @ Bw[N,K]^T (+bias).
// Block: 256 thr = 8 waves, computes a 128x128 C tile.
// Waves arranged 2(M-groups of 64) x 4(N-groups of 32); each wave owns
// 4x2 = 8 WMMA accumulators (64x32 output).
// Per K-step (32): stage A(128x32) and B(128x32) panels into LDS
// (async global->LDS on CDNA5), then 8 WMMAs per wave from LDS fragments.
// grid = (N/128, M/128).
// ---------------------------------------------------------------------------
template <bool OUTF32>
__global__ void gemm_bf16_wmma(const __bf16* __restrict__ A, int lda,
                               const __bf16* __restrict__ Bw, int ldb,
                               void* __restrict__ Cp, int ldc,
                               const float* __restrict__ bias, int K) {
    __shared__ __align__(16) __bf16 As[128][32];
    __shared__ __align__(16) __bf16 Bs[128][32];

    const int t    = threadIdx.x;
    const int lane = t & 31;
    const int wave = t >> 5;
    const int wm   = wave >> 2;        // 0..1  -> 64-row group
    const int wn   = wave & 3;         // 0..3  -> 32-col group
    const int l15  = lane & 15;

    const int m0 = blockIdx.y * 128;
    const int n0 = blockIdx.x * 128;

    v8f acc[4][2];
#pragma unroll
    for (int mt = 0; mt < 4; ++mt)
#pragma unroll
        for (int nt = 0; nt < 2; ++nt) acc[mt][nt] = v8f{};

    for (int k0 = 0; k0 < K; k0 += 32) {
        __syncthreads();  // previous iteration's LDS reads are done
        // stage 8KB A-panel + 8KB B-panel: 512 16B segments each,
        // 2 per thread per panel. seg -> (row = seg/4, col = (seg%4)*8)
#pragma unroll
        for (int s = 0; s < 2; ++s) {
            const int seg = t + s * 256;
            const int row = seg >> 2;
            const int col = (seg & 3) * 8;
            stage16(A + (size_t)(m0 + row) * lda + k0 + col, &As[row][col]);
            stage16(Bw + (size_t)(n0 + row) * ldb + k0 + col, &Bs[row][col]);
        }
        wait_stage();
        __syncthreads();  // panels visible to all waves

        v16bf bfr[2];
#pragma unroll
        for (int nt = 0; nt < 2; ++nt)
            bfr[nt] = load_b_frag(&Bs[wn * 32 + nt * 16 + l15][0], lane);
#pragma unroll
        for (int mt = 0; mt < 4; ++mt) {
            v16bf a = load_a_frag(&As[wm * 64 + mt * 16 + l15][0], lane);
#pragma unroll
            for (int nt = 0; nt < 2; ++nt)
                acc[mt][nt] = wmma_bf16(a, bfr[nt], acc[mt][nt]);
        }
    }

    // C/D layout: lanes 0-15 hold rows 0..7 (vgpr j), lanes 16-31 rows 8..15.
    const int rsel = (lane < 16) ? 0 : 8;
#pragma unroll
    for (int mt = 0; mt < 4; ++mt) {
        const int mrow = m0 + wm * 64 + mt * 16 + rsel;
#pragma unroll
        for (int nt = 0; nt < 2; ++nt) {
            const int col = n0 + wn * 32 + nt * 16 + l15;
            const float bv = (OUTF32 && bias) ? bias[col] : 0.0f;
#pragma unroll
            for (int j = 0; j < 8; ++j) {
                const float v = acc[mt][nt][j] + bv;
                const size_t idx = (size_t)(mrow + j) * ldc + col;
                if (OUTF32) reinterpret_cast<float*>(Cp)[idx] = v;
                else        reinterpret_cast<__bf16*>(Cp)[idx] = (__bf16)v;
            }
        }
    }
}

// ---------------------------------------------------------------------------
// RoPE rotation + layout reorg.
//   q_rot/k_rot : [b,h,c,dh]  (row-major dh — A/B fragment friendly)
//   vT          : [b,h,dh,c]  (keys contiguous — PV B-fragment friendly)
// theta_i = 10000^(-2*(i-1)/dh)  (faithful (i-1) quirk of the reference)
// ---------------------------------------------------------------------------
__global__ void rope_reorg(const __bf16* __restrict__ qkv,  // [ROWS, 3072]
                           __bf16* __restrict__ qrot,
                           __bf16* __restrict__ krot,
                           __bf16* __restrict__ vT) {
    int idx = blockIdx.x * blockDim.x + threadIdx.x;  // B*H*C*(DH/2) threads
    const int pairs = DH / 2;
    if (idx >= BB * HH * CC * pairs) return;

    const int i = idx % pairs;
    const int c = (idx / pairs) % CC;
    const int h = (idx / (pairs * CC)) % HH;
    const int b = idx / (pairs * CC * HH);
    const int bh = b * HH + h;

    const __bf16* row = qkv + ((size_t)(b * CC + c)) * QKVN + h * DH;

    const float theta = __powf(10000.0f, -2.0f * ((float)i - 1.0f) / (float)DH);
    float s, co;
    __sincosf((float)c * theta, &s, &co);

    // q rotation
    {
        const float ev = (float)row[2 * i], od = (float)row[2 * i + 1];
        __bf16* dst = qrot + ((size_t)bh * CC + c) * DH;
        dst[2 * i]     = (__bf16)(ev * co + od * s);
        dst[2 * i + 1] = (__bf16)(-ev * s + od * co);
    }
    // k rotation
    {
        const float ev = (float)row[DM + 2 * i], od = (float)row[DM + 2 * i + 1];
        __bf16* dst = krot + ((size_t)bh * CC + c) * DH;
        dst[2 * i]     = (__bf16)(ev * co + od * s);
        dst[2 * i + 1] = (__bf16)(-ev * s + od * co);
    }
    // v transpose
    {
        vT[((size_t)bh * DH + 2 * i) * CC + c]     = row[2 * DM + 2 * i];
        vT[((size_t)bh * DH + 2 * i + 1) * CC + c] = row[2 * DM + 2 * i + 1];
    }
}

// ---------------------------------------------------------------------------
// Attention. One wave per (b,h, 16-query tile). Block = 8 waves.
// Pass 1: den[m] = sum over ALL 2048 keys of exp(q.k/8)  (reference quirk:
//         denominator is unmasked and uses UNrotated q,k straight from qkv).
// Pass 2: P = exp(q_rot.k_rot/8) causally masked, staged 16x32 in LDS,
//         reloaded as an A-fragment, PV via WMMA into 16x64 accumulators.
// ---------------------------------------------------------------------------
__global__ void attn_wmma(const __bf16* __restrict__ qkv,   // [ROWS, 3072]
                          const __bf16* __restrict__ qrot,  // [b,h,c,dh]
                          const __bf16* __restrict__ krot,  // [b,h,c,dh]
                          const __bf16* __restrict__ vT,    // [b,h,dh,c]
                          __bf16* __restrict__ y) {         // [ROWS, DM] head-major cols
    __shared__ __align__(16) __bf16 pbuf[8][16][32];

    const int lane = threadIdx.x & 31;
    const int wave = threadIdx.x >> 5;
    const int gw   = blockIdx.x * 8 + wave;          // 4096 waves total
    const int qt   = gw % (CC / 16);                 // query tile 0..127
    const int bh   = gw / (CC / 16);
    const int b    = bh / HH;
    const int h    = bh % HH;

    const int l15  = lane & 15;
    const int rsel = (lane < 16) ? 0 : 8;            // C-layout row group

    // ---- preload A-fragments: rotated q (num) and raw q (den), K = dh = 64
    v16bf qr0, qr1, qd0, qd1;
    {
        const int r = qt * 16 + l15;
        const __bf16* qrp = qrot + ((size_t)bh * CC + r) * DH;
        qr0 = load_a_frag(qrp, lane);
        qr1 = load_a_frag(qrp + 32, lane);
        const __bf16* qdp = qkv + ((size_t)(b * CC + r)) * QKVN + h * DH;
        qd0 = load_a_frag(qdp, lane);
        qd1 = load_a_frag(qdp + 32, lane);
    }

    // ---- pass 1: unmasked denominator over all keys
    v8f den = v8f{};
    for (int kt = 0; kt < CC / 16; ++kt) {
        const int n = kt * 16 + l15;
        const __bf16* kp = qkv + ((size_t)(b * CC + n)) * QKVN + DM + h * DH;
        v16bf kb0 = load_b_frag(kp, lane);
        v16bf kb1 = load_b_frag(kp + 32, lane);
        v8f s = v8f{};
        s = wmma_bf16(qd0, kb0, s);
        s = wmma_bf16(qd1, kb1, s);
#pragma unroll
        for (int j = 0; j < 8; ++j) den[j] += __expf(s[j] * 0.125f);
    }
    // row-sum over the 16 columns held by each half-wave group
#pragma unroll
    for (int m = 1; m <= 8; m <<= 1) {
#pragma unroll
        for (int j = 0; j < 8; ++j) den[j] += __shfl_xor(den[j], m, 32);
    }

    // ---- pass 2: masked numerator + PV
    v8f accY[4] = {v8f{}, v8f{}, v8f{}, v8f{}};
    for (int kt = 0; kt <= qt; kt += 2) {
#pragma unroll
        for (int s2 = 0; s2 < 2; ++s2) {
            const int ktile = kt + s2;
            if (ktile <= qt) {
                const int n = ktile * 16 + l15;
                const __bf16* kp = krot + ((size_t)bh * CC + n) * DH;
                v16bf kb0 = load_b_frag(kp, lane);
                v16bf kb1 = load_b_frag(kp + 32, lane);
                v8f sn = v8f{};
                sn = wmma_bf16(qr0, kb0, sn);
                sn = wmma_bf16(qr1, kb1, sn);
#pragma unroll
                for (int j = 0; j < 8; ++j) {
                    const int mrow = qt * 16 + rsel + j;
                    const int kcol = ktile * 16 + l15;
                    const float p = (kcol <= mrow) ? __expf(sn[j] * 0.125f) : 0.0f;
                    pbuf[wave][rsel + j][s2 * 16 + l15] = (__bf16)p;
                }
            } else {
#pragma unroll
                for (int j = 0; j < 8; ++j)
                    pbuf[wave][rsel + j][s2 * 16 + l15] = (__bf16)0.0f;
            }
        }
        // wave-local LDS write->read ordering (split counters on CDNA5)
        asm volatile("s_wait_dscnt 0x0" ::: "memory");

        // reload P as an A-fragment (16x32 bf16)
        FragAB pa;
        {
            const int off = (lane < 16) ? 0 : 8;
            pa.u[0] = *reinterpret_cast<const uint4*>(&pbuf[wave][l15][off]);
            pa.u[1] = *reinterpret_cast<const uint4*>(&pbuf[wave][l15][off + 16]);
        }
        asm volatile("s_wait_dscnt 0x0" ::: "memory");

#pragma unroll
        for (int nt = 0; nt < 4; ++nt) {
            const int d = nt * 16 + l15;
            const __bf16* vp = vT + ((size_t)bh * DH + d) * CC + kt * 16;
            v16bf vb = load_b_frag(vp, lane);
            accY[nt] = wmma_bf16(pa.v, vb, accY[nt]);
        }
    }

    // ---- divide by denominator, store y (bf16, head-major columns)
    const int mrow = qt * 16 + rsel;
#pragma unroll
    for (int nt = 0; nt < 4; ++nt) {
#pragma unroll
        for (int j = 0; j < 8; ++j) {
            const float v = accY[nt][j] / den[j];
            const size_t r = (size_t)(b * CC + mrow + j);
            y[r * DM + h * DH + nt * 16 + l15] = (__bf16)v;
        }
    }
}

// ---------------------------------------------------------------------------
// Host-side launch
// ---------------------------------------------------------------------------
extern "C" void kernel_launch(void* const* d_in, const int* in_sizes, int n_in,
                              void* d_out, int out_size, void* d_ws, size_t ws_size,
                              hipStream_t stream) {
    const float* x     = reinterpret_cast<const float*>(d_in[0]); // [B,C,D]
    const float* w_qkv = reinterpret_cast<const float*>(d_in[1]); // [3D,D]
    const float* w_out = reinterpret_cast<const float*>(d_in[2]); // [D,D]
    const float* b_out = reinterpret_cast<const float*>(d_in[3]); // [D]
    float* out = reinterpret_cast<float*>(d_out);

    // workspace carve-up (bf16 buffers)
    char* w = reinterpret_cast<char*>(d_ws);
    auto take = [&](size_t elems) {
        __bf16* p = reinterpret_cast<__bf16*>(w);
        w += elems * sizeof(__bf16);
        return p;
    };
    __bf16* xb    = take((size_t)ROWS * DM);      // x bf16
    __bf16* wqkvb = take((size_t)QKVN * DM);      // w_qkv bf16
    __bf16* woutb = take((size_t)DM * DM);        // w_out bf16
    __bf16* qkvb  = take((size_t)ROWS * QKVN);    // qkv projections
    __bf16* qrot  = take((size_t)BB * HH * CC * DH);
    __bf16* krot  = take((size_t)BB * HH * CC * DH);
    __bf16* vTb   = take((size_t)BB * HH * DH * CC);
    __bf16* yb    = take((size_t)ROWS * DM);      // attention output

    // 1) casts to bf16
    {
        int n;
        n = ROWS * DM;
        cast_f32_to_bf16<<<(n + 255) / 256, 256, 0, stream>>>(x, xb, n);
        n = QKVN * DM;
        cast_f32_to_bf16<<<(n + 255) / 256, 256, 0, stream>>>(w_qkv, wqkvb, n);
        n = DM * DM;
        cast_f32_to_bf16<<<(n + 255) / 256, 256, 0, stream>>>(w_out, woutb, n);
    }

    // 2) QKV projection: qkv = x @ w_qkv^T   (M=4096, N=3072, K=1024)
    gemm_bf16_wmma<false><<<dim3(QKVN / 128, ROWS / 128), 256, 0, stream>>>(
        xb, DM, wqkvb, DM, qkvb, QKVN, nullptr, DM);

    // 3) RoPE rotation + v transpose
    {
        int n = BB * HH * CC * (DH / 2);
        rope_reorg<<<(n + 255) / 256, 256, 0, stream>>>(qkvb, qrot, krot, vTb);
    }

    // 4) attention (one wave per 16-query tile per head)
    {
        int waves = BB * HH * (CC / 16);  // 4096
        attn_wmma<<<waves / 8, 256, 0, stream>>>(qkvb, qrot, krot, vTb, yb);
    }

    // 5) output projection: out = y @ w_out^T + b_out  (M=4096, N=1024, K=1024)
    gemm_bf16_wmma<true><<<dim3(DM / 128, ROWS / 128), 256, 0, stream>>>(
        yb, DM, woutb, DM, out, DM, b_out, DM);
}
